// MultiHead_26173530702584
// MI455X (gfx1250) — matrix operand order repaired and checked by reading, not compiled, not run on hardware
//
#include <hip/hip_runtime.h>
#include <hip/hip_bf16.h>

// ---------------------------------------------------------------------------
// Multi-head attention for MI455X (gfx1250), wave32 + WMMA bf16.
//   B=4, S=1024, D=1024, H=16, HD=64.
// Pipeline:
//   1) wtrans: cast fp32 weights -> bf16, transposed [N][K] (B-fragment layout)
//   2) gemm128<0..2>: Q/K/V projections (bf16 WMMA, f32 accum, bias, Q pre-scaled)
//        Q -> [BH][S][64], K -> [BH][S][64], V -> [BH][64][S] (transposed)
//      BK=64, double-buffered LDS, async global->LDS staging (ASYNCcnt),
//      coalesced b128 epilogue via LDS tile.
//   3) flash_attn: online-softmax attention, 64-key blocks, double-buffered
//      K/V tiles staged with global_load_async_to_lds_b128, WMMA for QK^T & PV.
//   4) gemm128<3>: output projection -> fp32 d_out
// ---------------------------------------------------------------------------

typedef unsigned short u16;
typedef __attribute__((ext_vector_type(16))) __bf16 v16bf;
typedef __attribute__((ext_vector_type(8)))  float  v8f;

union FragBF { v16bf v; uint4 q[2]; };

__device__ __forceinline__ u16 f2bf(float f) {
  union { float f; unsigned int u; } x; x.f = f;
  unsigned int r = 0x7FFFu + ((x.u >> 16) & 1u);   // round-to-nearest-even
  return (u16)((x.u + r) >> 16);
}

__device__ __forceinline__ v8f zero8() {
  v8f z = {0.f,0.f,0.f,0.f,0.f,0.f,0.f,0.f};
  return z;
}

// Async 16B global -> LDS copy (CDNA5 GLOBAL_LOAD_ASYNC_TO_LDS_B128, ASYNCcnt).
// LDS operand is the low 32 bits of the generic pointer.
__device__ __forceinline__ void async_cp16(const void* gaddr, void* lds) {
  unsigned ldsoff = (unsigned)(unsigned long long)lds;
  asm volatile("global_load_async_to_lds_b128 %0, %1, off"
               :: "v"(ldsoff), "v"(gaddr) : "memory");
}
__device__ __forceinline__ void async_wait0() {
  asm volatile("s_wait_asynccnt 0" ::: "memory");
}

// ---------------------------------------------------------------------------
// Weight cast + transpose: w fp32 [K=1024][N=1024] -> wt bf16 [N][K]
// ---------------------------------------------------------------------------
__global__ __launch_bounds__(256) void wtrans(const float* __restrict__ w,
                                              u16* __restrict__ wt) {
  __shared__ float t[32][33];
  const int k0 = blockIdx.y * 32, n0 = blockIdx.x * 32;
  for (int i = threadIdx.y; i < 32; i += 8)
    t[i][threadIdx.x] = w[(size_t)(k0 + i) * 1024 + n0 + threadIdx.x];
  __syncthreads();
  for (int i = threadIdx.y; i < 32; i += 8)
    wt[(size_t)(n0 + i) * 1024 + k0 + threadIdx.x] = f2bf(t[threadIdx.x][i]);
}

// ---------------------------------------------------------------------------
// 128x128-tile GEMM, M=4096, N=1024, K=1024, BK=64, double-buffered LDS.
//   A: MODE<3 fp32 row-major (cast to bf16 while staging), MODE==3 bf16 (async).
//   B: bf16 transposed [N][K] (natural B-fragment layout), async staged.
// MODE 0: Q out  -> bf16 [BH][S][64], scaled by 1/sqrt(64), +bias
// MODE 1: K out  -> bf16 [BH][S][64], +bias
// MODE 2: V out  -> bf16 [BH][64][S] (transposed), +bias
// MODE 3: fp32 out [M][N], no bias
// ---------------------------------------------------------------------------
template <int MODE>
__global__ __launch_bounds__(256) void gemm128(const void* __restrict__ Ap,
                                               const u16* __restrict__ Bt,
                                               const float* __restrict__ bias,
                                               void* __restrict__ Outp) {
  // [buf][128 rows][64 K + 8 pad];  epilogue reuses As as Ct[128][136]
  __shared__ __align__(16) u16 As[2][128 * 72];
  __shared__ __align__(16) u16 Bs[2][128 * 72];

  const int tid  = threadIdx.x;
  const int lane = tid & 31, wave = tid >> 5;
  const int lh = lane >> 4, ln = lane & 15;
  const int m0 = blockIdx.y * 128;
  const int n0 = blockIdx.x * 128;
  const int waveM = wave & 3;    // 4 row groups of 32
  const int waveN = wave >> 2;   // 2 col groups of 64

  v8f acc[2][4];
#pragma unroll
  for (int i = 0; i < 2; ++i)
#pragma unroll
    for (int j = 0; j < 4; ++j) acc[i][j] = zero8();

  // ---- stage one 64-wide K slab into buffer `buf` ----
  auto stage = [&](int buf, int k0) {
    if (MODE < 3) {  // fp32 A -> bf16 LDS (convert in flight, register route)
      const float* A = (const float*)Ap;
#pragma unroll
      for (int i = 0; i < 8; ++i) {
        int c = tid + i * 256;               // 2048 float4 chunks
        int row = c >> 4, col = (c & 15) * 4;
        float4 vv = *(const float4*)(A + (size_t)(m0 + row) * 1024 + k0 + col);
        union { u16 u[4]; uint2 d; } p;
        p.u[0] = f2bf(vv.x); p.u[1] = f2bf(vv.y);
        p.u[2] = f2bf(vv.z); p.u[3] = f2bf(vv.w);
        *(uint2*)&As[buf][row * 72 + col] = p.d;
      }
    } else {         // bf16 A: async global -> LDS
      const u16* A = (const u16*)Ap;
#pragma unroll
      for (int i = 0; i < 4; ++i) {
        int c = tid + i * 256;               // 1024 uint4 chunks
        int row = c >> 3, col = (c & 7) * 8;
        async_cp16(A + (size_t)(m0 + row) * 1024 + k0 + col,
                   &As[buf][row * 72 + col]);
      }
    }
#pragma unroll
    for (int i = 0; i < 4; ++i) {
      int c = tid + i * 256;
      int row = c >> 3, col = (c & 7) * 8;
      async_cp16(Bt + (size_t)(n0 + row) * 1024 + k0 + col,
                 &Bs[buf][row * 72 + col]);
    }
  };

  stage(0, 0);
  async_wait0();
  __syncthreads();

  for (int k0 = 0; k0 < 1024; k0 += 64) {
    const int cur = (k0 >> 6) & 1;
    if (k0 < 960) stage(cur ^ 1, k0 + 64);  // async-prefetch next slab

#pragma unroll
    for (int ks = 0; ks < 2; ++ks) {        // two 32-wide WMMA K-steps
      // A fragment (ISA 16x32 bf16): halves 0-7 at K=lh*8, 8-15 at K=16+lh*8
      FragBF af[2];
#pragma unroll
      for (int mt = 0; mt < 2; ++mt) {
        int r = waveM * 32 + mt * 16 + ln;
        af[mt].q[0] = *(const uint4*)&As[cur][r * 72 + ks * 32 + lh * 8];
        af[mt].q[1] = *(const uint4*)&As[cur][r * 72 + ks * 32 + 16 + lh * 8];
      }
      // B fragment: col = ln, halves 0-7 at K=lh*8, 8-15 at K=16+lh*8 (matches A)
#pragma unroll
      for (int nt = 0; nt < 4; ++nt) {
        FragBF bf;
        int r = waveN * 64 + nt * 16 + ln;
        bf.q[0] = *(const uint4*)&Bs[cur][r * 72 + ks * 32 + lh * 8];
        bf.q[1] = *(const uint4*)&Bs[cur][r * 72 + ks * 32 + 16 + lh * 8];
#pragma unroll
        for (int mt = 0; mt < 2; ++mt)
          acc[mt][nt] = __builtin_amdgcn_wmma_f32_16x16x32_bf16(
              false, af[mt].v, false, bf.v, (short)0, acc[mt][nt], false, false);
      }
    }
    async_wait0();       // next slab's async copies landed in LDS
    __syncthreads();
  }

  // ---- epilogue.  C layout: VGPR g, lane -> (M = g + 8*lh, N = ln). ----
  if (MODE == 3) {
#pragma unroll
    for (int mt = 0; mt < 2; ++mt)
#pragma unroll
      for (int nt = 0; nt < 4; ++nt) {
        int n = n0 + waveN * 64 + nt * 16 + ln;
#pragma unroll
        for (int g = 0; g < 8; ++g) {
          int m = m0 + waveM * 32 + mt * 16 + g + 8 * lh;
          ((float*)Outp)[(size_t)m * 1024 + n] = acc[mt][nt][g];
        }
      }
    return;
  }

  // bf16 outputs: go through LDS tile for coalesced 16B global stores.
  u16* Ct = &As[0][0];  // reuse staging LDS: [128][136] (17,408 u16 <= 18,432)
#pragma unroll
  for (int mt = 0; mt < 2; ++mt)
#pragma unroll
    for (int nt = 0; nt < 4; ++nt) {
      int nloc = waveN * 64 + nt * 16 + ln;
      float bv = bias[n0 + nloc];
#pragma unroll
      for (int g = 0; g < 8; ++g) {
        int mloc = waveM * 32 + mt * 16 + g + 8 * lh;
        float val = acc[mt][nt][g] + bv;
        if (MODE == 0) val *= 0.125f;                 // 1/sqrt(HD=64) into Q
        if (MODE == 2) Ct[nloc * 136 + mloc] = f2bf(val);  // transposed tile
        else           Ct[mloc * 136 + nloc] = f2bf(val);
      }
    }
  __syncthreads();

  u16* O = (u16*)Outp;
#pragma unroll
  for (int i = 0; i < 8; ++i) {               // 2048 uint4 chunks, 8/thread
    int c = tid + i * 256;
    int rloc = c >> 4, cch = (c & 15) * 8;
    uint4 val = *(const uint4*)&Ct[rloc * 136 + cch];
    if (MODE == 2) {
      int n = n0 + rloc, h = n >> 6, hd = n & 63;
      int m = m0 + cch,  b = m >> 10, s = m & 1023;
      *(uint4*)&O[(((size_t)(b * 16 + h) * 64 + hd) * 1024 + s)] = val;  // V^T
    } else {
      int m = m0 + rloc, b = m >> 10, s = m & 1023;
      int n = n0 + cch,  h = n >> 6,  hd = n & 63;
      *(uint4*)&O[(((size_t)(b * 16 + h) * 1024 + s) * 64 + hd)] = val;
    }
  }
}

// ---------------------------------------------------------------------------
// Flash attention. Grid: (S/128, B*H). 8 waves; wave owns 16 query rows,
// streams keys in blocks of 64 with online softmax; K/V tiles double-buffered
// and staged with async global->LDS copies.
// ---------------------------------------------------------------------------
__global__ __launch_bounds__(256) void flash_attn(const u16* __restrict__ Qh,
                                                  const u16* __restrict__ Kh,
                                                  const u16* __restrict__ Vt,
                                                  const float* __restrict__ mask,
                                                  u16* __restrict__ Ctx) {
  __shared__ __align__(16) u16 Ks[2][64 * 72];  // [64 keys][64 hd + 8 pad]
  __shared__ __align__(16) u16 Vs[2][64 * 72];  // [64 hd][64 keys + 8 pad] V^T
  __shared__ __align__(16) u16 Ps[8][16 * 72];  // per-wave P [16][64 + 8 pad]

  const int tid = threadIdx.x;
  const int lane = tid & 31, wave = tid >> 5;
  const int lh = lane >> 4, ln = lane & 15;
  const int bh = blockIdx.y;
  const int b = bh >> 4, h = bh & 15;
  const int qbase = blockIdx.x * 128 + wave * 16;

  // Q fragments (16 rows x 64 hd = two 16x32 A-fragments), loaded once.
  const u16* Qrow = Qh + ((size_t)bh * 1024 + qbase + ln) * 64;
  FragBF qf[2];
  qf[0].q[0] = *(const uint4*)(Qrow + lh * 8);
  qf[0].q[1] = *(const uint4*)(Qrow + 16 + lh * 8);
  qf[1].q[0] = *(const uint4*)(Qrow + 32 + lh * 8);
  qf[1].q[1] = *(const uint4*)(Qrow + 48 + lh * 8);

  v8f acc[4];
#pragma unroll
  for (int t = 0; t < 4; ++t) acc[t] = zero8();
  float mrow[8], lrow[8];
#pragma unroll
  for (int g = 0; g < 8; ++g) { mrow[g] = -1e30f; lrow[g] = 0.f; }

  const float* mb = mask + (size_t)b * 1024 * 1024;

  auto stageKV = [&](int buf, int kb) {
#pragma unroll
    for (int i = 0; i < 2; ++i) {             // 512 uint4 chunks each
      int c = tid + i * 256;
      int row = c >> 3, col = (c & 7) * 8;
      async_cp16(Kh + ((size_t)bh * 1024 + kb + row) * 64 + col,
                 &Ks[buf][row * 72 + col]);
      async_cp16(Vt + ((size_t)bh * 64 + row) * 1024 + kb + col,
                 &Vs[buf][row * 72 + col]);
    }
  };

  stageKV(0, 0);
  async_wait0();
  __syncthreads();

  for (int kb = 0; kb < 1024; kb += 64) {
    const int cur = (kb >> 6) & 1;
    if (kb < 960) stageKV(cur ^ 1, kb + 64);   // async-prefetch next block

    // scores = (Q*scale) @ K^T : 4 tiles of 16 keys (scale folded into Q)
    v8f sc[4];
#pragma unroll
    for (int nt = 0; nt < 4; ++nt) {
      v8f c = zero8();
#pragma unroll
      for (int kh = 0; kh < 2; ++kh) {
        FragBF bf;
        int key = nt * 16 + ln;
        bf.q[0] = *(const uint4*)&Ks[cur][key * 72 + kh * 32 + lh * 8];
        bf.q[1] = *(const uint4*)&Ks[cur][key * 72 + kh * 32 + 16 + lh * 8];
        c = __builtin_amdgcn_wmma_f32_16x16x32_bf16(false, qf[kh].v, false,
                                                    bf.v, (short)0, c, false,
                                                    false);
      }
      sc[nt] = c;
    }
    // additive mask (reference: score += mask * -1e9)
#pragma unroll
    for (int g = 0; g < 8; ++g) {
      int sq = qbase + g + 8 * lh;
#pragma unroll
      for (int nt = 0; nt < 4; ++nt)
        sc[nt][g] += mb[(size_t)sq * 1024 + kb + nt * 16 + ln] * -1e9f;
    }
    // online softmax: row reductions within 16-lane halves
    float alpha[8];
#pragma unroll
    for (int g = 0; g < 8; ++g) {
      float x = fmaxf(fmaxf(sc[0][g], sc[1][g]), fmaxf(sc[2][g], sc[3][g]));
      x = fmaxf(x, __shfl_xor(x, 1, 32));
      x = fmaxf(x, __shfl_xor(x, 2, 32));
      x = fmaxf(x, __shfl_xor(x, 4, 32));
      x = fmaxf(x, __shfl_xor(x, 8, 32));
      float mn = fmaxf(mrow[g], x);
      alpha[g] = __expf(mrow[g] - mn);
      mrow[g] = mn;
      float s = 0.f;
#pragma unroll
      for (int nt = 0; nt < 4; ++nt) {
        float p = __expf(sc[nt][g] - mn);
        sc[nt][g] = p;
        s += p;
      }
      s += __shfl_xor(s, 1, 32);
      s += __shfl_xor(s, 2, 32);
      s += __shfl_xor(s, 4, 32);
      s += __shfl_xor(s, 8, 32);
      lrow[g] = lrow[g] * alpha[g] + s;
    }
#pragma unroll
    for (int t = 0; t < 4; ++t)
#pragma unroll
      for (int g = 0; g < 8; ++g) acc[t][g] *= alpha[g];

    // C-layout P -> A-layout via per-wave LDS tile (same-wave LDS is in-order)
    u16* P = Ps[wave];
#pragma unroll
    for (int g = 0; g < 8; ++g) {
      int r = g + 8 * lh;
#pragma unroll
      for (int nt = 0; nt < 4; ++nt)
        P[r * 72 + nt * 16 + ln] = f2bf(sc[nt][g]);
    }
    FragBF pf[2];
#pragma unroll
    for (int kh = 0; kh < 2; ++kh) {
      pf[kh].q[0] = *(const uint4*)&P[ln * 72 + kh * 32 + lh * 8];
      pf[kh].q[1] = *(const uint4*)&P[ln * 72 + kh * 32 + 16 + lh * 8];
    }
    // acc += P @ V  (B-fragment from V^T tile)
#pragma unroll
    for (int t = 0; t < 4; ++t) {
#pragma unroll
      for (int kh = 0; kh < 2; ++kh) {
        FragBF bf;
        int hd = t * 16 + ln;
        bf.q[0] = *(const uint4*)&Vs[cur][hd * 72 + kh * 32 + lh * 8];
        bf.q[1] = *(const uint4*)&Vs[cur][hd * 72 + kh * 32 + 16 + lh * 8];
        acc[t] = __builtin_amdgcn_wmma_f32_16x16x32_bf16(
            false, pf[kh].v, false, bf.v, (short)0, acc[t], false, false);
      }
    }
    async_wait0();       // next block's K/V tiles landed in LDS
    __syncthreads();
  }

  // ctx[b][s][h*64+hd] = acc / l   (bf16, feeds output-projection GEMM)
#pragma unroll
  for (int g = 0; g < 8; ++g) {
    float rl = 1.0f / lrow[g];
    int sq = qbase + g + 8 * lh;
    size_t base = ((size_t)b * 1024 + sq) * 1024 + (size_t)h * 64;
#pragma unroll
    for (int t = 0; t < 4; ++t)
      Ctx[base + t * 16 + ln] = f2bf(acc[t][g] * rl);
  }
}

// ---------------------------------------------------------------------------
extern "C" void kernel_launch(void* const* d_in, const int* in_sizes, int n_in,
                              void* d_out, int out_size, void* d_ws,
                              size_t ws_size, hipStream_t stream) {
  const float* q    = (const float*)d_in[0];
  const float* k    = (const float*)d_in[1];
  const float* v    = (const float*)d_in[2];
  const float* mask = (const float*)d_in[3];
  const float* wq   = (const float*)d_in[4];
  const float* bq   = (const float*)d_in[5];
  const float* wk   = (const float*)d_in[6];
  const float* bk   = (const float*)d_in[7];
  const float* wv   = (const float*)d_in[8];
  const float* bv   = (const float*)d_in[9];
  const float* wo   = (const float*)d_in[10];
  float* out = (float*)d_out;

  char* ws = (char*)d_ws;
  u16* WqT = (u16*)(ws + ((size_t)0 << 20));   // 2 MB each, bf16 [N][K]
  u16* WkT = (u16*)(ws + ((size_t)2 << 20));
  u16* WvT = (u16*)(ws + ((size_t)4 << 20));
  u16* WoT = (u16*)(ws + ((size_t)6 << 20));
  u16* Qh  = (u16*)(ws + ((size_t)8 << 20));   // 8 MB: [BH][S][64] bf16
  u16* Kh  = (u16*)(ws + ((size_t)16 << 20));  // 8 MB: [BH][S][64]
  u16* Vt  = (u16*)(ws + ((size_t)24 << 20));  // 8 MB: [BH][64][S]
  u16* Ctx = (u16*)(ws + ((size_t)32 << 20));  // 8 MB: [B*S][D]

  dim3 tb(32, 8), tg(32, 32);
  wtrans<<<tg, tb, 0, stream>>>(wq, WqT);
  wtrans<<<tg, tb, 0, stream>>>(wk, WkT);
  wtrans<<<tg, tb, 0, stream>>>(wv, WvT);
  wtrans<<<tg, tb, 0, stream>>>(wo, WoT);

  dim3 gg(8, 32);  // N/128, M/128
  gemm128<0><<<gg, 256, 0, stream>>>(q, WqT, bq, Qh);
  gemm128<1><<<gg, 256, 0, stream>>>(k, WkT, bk, Kh);
  gemm128<2><<<gg, 256, 0, stream>>>(v, WvT, bv, Vt);

  flash_attn<<<dim3(8, 64), 256, 0, stream>>>(Qh, Kh, Vt, mask, Ctx);

  gemm128<3><<<gg, 256, 0, stream>>>(Ctx, WoT, nullptr, out);
}
